// FeatureMagnet_60172491817901
// MI455X (gfx1250) — matrix operand order repaired and checked by reading, not compiled, not run on hardware
//
#include <hip/hip_runtime.h>

// ---------------- constants ----------------
#define Cdim 256
#define Hh   8
#define DH   32
#define DF   1024
#define LE   2048
#define LA   2048
#define LQ   3072   // LE + n_remain (1024)
#define LK   4096   // LE + LA

typedef _Float16 h16;
typedef __attribute__((ext_vector_type(16))) _Float16 v16h;
typedef __attribute__((ext_vector_type(8)))  float    v8f;

// ---------------- WMMA fragment loaders (16x16x32 f16, this file's contiguous k order k = 16*hg + i) ----------------
// A 16x32 f16: lanes 0-15 row=lane, elems 0..15 =K0..15
//              lanes 16-31 row=lane-16, elems 0..15 =K16..31
__device__ __forceinline__ v16h load_afrag(const h16* base, int stride) {
    int lane = threadIdx.x & 31;
    int hg   = lane >> 4;
    int r    = lane & 15;
    const h16* row = base + (size_t)r * stride;
    int k0 = hg * 16;
    v16h a;
#pragma unroll
    for (int i = 0; i < 16; ++i) a[i] = row[k0 + i];
    return a;
}

// B 32x16 where B[k][n] = W[n][k] (W row-major, stride ldw). lanes 0-15: K=0..15,
// lanes 16-31: K=16..31; col = lane&15.  w points at W[nbase][kbase].
__device__ __forceinline__ v16h load_bfrag_nt(const h16* w, int stride) {
    int lane = threadIdx.x & 31;
    int hg   = lane >> 4;
    int col  = lane & 15;
    const h16* row = w + (size_t)col * stride + hg * 16;
    v16h b;
#pragma unroll
    for (int i = 0; i < 16; ++i) b[i] = row[i];
    return b;
}

// B 32x16 from row-major B[K][N] (e.g. V tile): b[i] = B[hg*16+i][col]
__device__ __forceinline__ v16h load_bfrag_nn(const h16* b0, int stride) {
    int lane = threadIdx.x & 31;
    int hg   = lane >> 4;
    int col  = lane & 15;
    v16h b;
#pragma unroll
    for (int i = 0; i < 16; ++i) b[i] = b0[(size_t)(hg * 16 + i) * stride + col];
    return b;
}

__device__ __forceinline__ v8f wmma_f16(v16h a, v16h b, v8f c) {
    return __builtin_amdgcn_wmma_f32_16x16x32_f16(false, a, false, b, (short)0, c, false, false);
}

// async global->LDS 16B copy (CDNA5): VDST = per-lane LDS byte offset, VADDR = 64-bit global addr
__device__ __forceinline__ void async_ld128(unsigned lds_byte_off, const void* gptr) {
    unsigned long long ga = (unsigned long long)gptr;
    asm volatile("global_load_async_to_lds_b128 %0, %1, off"
                 :: "v"(lds_byte_off), "v"(ga) : "memory");
}
__device__ __forceinline__ void async_wait0() {
    asm volatile("s_wait_asynccnt 0x0" ::: "memory");
}

// ---------------- small utility kernels ----------------
__global__ void k_cvt(const float* __restrict__ s, h16* __restrict__ d, int n) {
    int i = blockIdx.x * blockDim.x + threadIdx.x;
    if (i < n) d[i] = (h16)s[i];
}

// For each ego token: find matching agent (first index), build fused_in = [x_ego, x_agent[idx]]
__global__ void k_match(const float* __restrict__ xe, const float* __restrict__ xa,
                        const int* __restrict__ pe, const int* __restrict__ pa,
                        int* __restrict__ a4e, h16* __restrict__ fused) {
    int i = blockIdx.x;
    __shared__ int sj;
    if (threadIdx.x == 0) sj = 0x7fffffff;
    __syncthreads();
    int p = pe[i];
    for (int j = threadIdx.x; j < LA; j += blockDim.x)
        if (pa[j] == p) atomicMin(&sj, j);
    __syncthreads();
    int jm = sj;
    int matched = (jm != 0x7fffffff);
    if (threadIdx.x == 0) a4e[i] = matched ? jm : -1;
    int jg = matched ? jm : 0;   // reference argmax -> 0 when no match
    for (int c = threadIdx.x; c < 2 * Cdim; c += blockDim.x) {
        float v = (c < Cdim) ? xe[(size_t)i * Cdim + c] : xa[(size_t)jg * Cdim + (c - Cdim)];
        fused[(size_t)i * (2 * Cdim) + c] = (h16)v;
    }
}

__global__ void k_matched_a(const int* __restrict__ pe, const int* __restrict__ pa,
                            int* __restrict__ ma) {
    int j = blockIdx.x * blockDim.x + threadIdx.x;
    if (j >= LA) return;
    int p = pa[j], m = 0;
    for (int i = 0; i < LE; ++i) m |= (pe[i] == p);
    ma[j] = m;
}

__global__ void k_remain(const int* __restrict__ ma, int* __restrict__ ridx) {
    int j = blockIdx.x * blockDim.x + threadIdx.x;
    if (j >= LA) return;
    if (ma[j]) return;
    int r = 0;
    for (int k = 0; k < j; ++k) r += (ma[k] == 0);
    ridx[r] = j;
}

__global__ void k_build_kv(const float* __restrict__ xe, const float* __restrict__ xa,
                           h16* __restrict__ kv) {
    int i = blockIdx.x * blockDim.x + threadIdx.x;
    if (i >= LK * Cdim) return;
    float v = (i < LE * Cdim) ? xe[i] : xa[i - LE * Cdim];
    kv[i] = (h16)v;
}

__global__ void k_build_q(const float* __restrict__ xe, const float* __restrict__ xa,
                          const float* __restrict__ fused3, const int* __restrict__ a4e,
                          const int* __restrict__ ridx,
                          float* __restrict__ qf, h16* __restrict__ qh) {
    int i = blockIdx.x * blockDim.x + threadIdx.x;
    if (i >= LQ * Cdim) return;
    int row = i / Cdim, c = i % Cdim;
    float v;
    if (row < LE) v = (a4e[row] >= 0) ? fused3[i] : xe[i];
    else          v = xa[(size_t)ridx[row - LE] * Cdim + c];
    qf[i] = v;
    qh[i] = (h16)v;
}

// LN(A+B) -> outf (f32) and optional outh (f16); blockDim == Cdim
__global__ void k_ln(const float* __restrict__ A, const float* __restrict__ B,
                     const float* __restrict__ g, const float* __restrict__ be,
                     float* __restrict__ outf, h16* __restrict__ outh) {
    int row = blockIdx.x, t = threadIdx.x;
    __shared__ float red[Cdim];
    float x = A[(size_t)row * Cdim + t] + B[(size_t)row * Cdim + t];
    red[t] = x; __syncthreads();
    for (int s = Cdim / 2; s > 0; s >>= 1) { if (t < s) red[t] += red[t + s]; __syncthreads(); }
    float mean = red[0] / Cdim; __syncthreads();
    float d = x - mean;
    red[t] = d * d; __syncthreads();
    for (int s = Cdim / 2; s > 0; s >>= 1) { if (t < s) red[t] += red[t + s]; __syncthreads(); }
    float var = red[0] / Cdim;
    float y = d * rsqrtf(var + 1e-5f) * g[t] + be[t];
    outf[(size_t)row * Cdim + t] = y;
    if (outh) outh[(size_t)row * Cdim + t] = (h16)y;
}

// ---------------- WMMA GEMM: C = act(A[M,K] @ W[N,K]^T + bias) ----------------
// BM=128, BN=64, BK=32. grid = (M/128, N/64), block = 256 (8 waves).
// wave -> (mi = w>>1, ni = w&1) computes a 32x32 block = 2x2 WMMA tiles.
// Dynamic LDS: At (128x32 f16) at byte 0, Wt (64x32 f16) at byte 8192. Total 12288.
#define GEMM_LDS_BYTES 12288
__global__ void k_gemm_nt(const h16* __restrict__ A, int lda,
                          const h16* __restrict__ W, int ldw,
                          const float* __restrict__ bias,
                          float* __restrict__ Cf, h16* __restrict__ Ch,
                          int ldc, int K, int act) {
    extern __shared__ char smem[];
    h16* At = (h16*)smem;
    h16* Wt = (h16*)(smem + 8192);
    int mb = blockIdx.x, nb = blockIdx.y;
    int t = threadIdx.x;
    int wv = t >> 5, lane = t & 31;
    int mi = wv >> 1, ni = wv & 1;
    int hg = lane >> 4, lcol = lane & 15;

    // staging coords: A tile 128x32 halves (32B/thread), W tile 64x32 halves (16B/thread)
    int ra = t >> 1, oa = (t & 1) * 16;
    int rw = t >> 2, ow = (t & 3) * 8;
    unsigned ldsA = (unsigned)((ra * 32 + oa) * 2);
    unsigned ldsW = (unsigned)(8192 + (rw * 32 + ow) * 2);

    v8f acc00 = {}, acc01 = {}, acc10 = {}, acc11 = {};
    for (int kt = 0; kt < K; kt += 32) {
        const h16* asrc = A + (size_t)(mb * 128 + ra) * lda + kt + oa;
        const h16* wsrc = W + (size_t)(nb * 64 + rw) * ldw + kt + ow;
        async_ld128(ldsA,       asrc);
        async_ld128(ldsA + 16u, asrc + 8);
        async_ld128(ldsW,       wsrc);
        if (kt + 32 < K) __builtin_prefetch(asrc + 32, 0, 1);
        async_wait0();
        __syncthreads();
        v16h a0 = load_afrag(At + (mi * 32) * 32, 32);
        v16h a1 = load_afrag(At + (mi * 32 + 16) * 32, 32);
        v16h b0 = load_bfrag_nt(Wt + (ni * 32) * 32, 32);
        v16h b1 = load_bfrag_nt(Wt + (ni * 32 + 16) * 32, 32);
        acc00 = wmma_f16(a0, b0, acc00);
        acc01 = wmma_f16(a0, b1, acc01);
        acc10 = wmma_f16(a1, b0, acc10);
        acc11 = wmma_f16(a1, b1, acc11);
        __syncthreads();
    }
    int gm0 = mb * 128 + mi * 32, gn0 = nb * 64 + ni * 32;
    auto store_tile = [&](v8f acc, int gmB, int gnB) {
#pragma unroll
        for (int v = 0; v < 8; ++v) {
            int gm = gmB + v + 8 * hg;
            int gn = gnB + lcol;
            float val = acc[v] + bias[gn];
            if (act) val = fmaxf(val, 0.f);
            if (Cf) Cf[(size_t)gm * ldc + gn] = val;
            if (Ch) Ch[(size_t)gm * ldc + gn] = (h16)val;
        }
    };
    store_tile(acc00, gm0,      gn0);
    store_tile(acc01, gm0,      gn0 + 16);
    store_tile(acc10, gm0 + 16, gn0);
    store_tile(acc11, gm0 + 16, gn0 + 16);
}

// ---------------- flash attention: per block 64 q-rows x one head ----------------
// Qp: [LQ,256] f16 (head h = cols h*32..). KVp: [LK,512] f16, K at col h*32, V at 256+h*32.
// Dynamic LDS: Kt 64x32 @0 (4KB), Vt 64x32 @4096 (4KB), Ps[4][16x64] @8192 (8KB). Total 16384.
#define ATTN_LDS_BYTES 16384
__global__ void k_attn(const h16* __restrict__ Qp, const h16* __restrict__ KVp,
                       h16* __restrict__ Oh) {
    extern __shared__ char smem[];
    h16* Kt = (h16*)smem;
    h16* Vt = (h16*)(smem + 4096);
    int qb = blockIdx.x * 64;
    int h  = blockIdx.y;
    int t = threadIdx.x;          // 128 threads = 4 waves
    int wv = t >> 5, lane = t & 31;
    int hg = lane >> 4, lcol = lane & 15;
    h16* ps = (h16*)(smem + 8192 + wv * 2048);

    v16h qf = load_afrag(Qp + (size_t)(qb + wv * 16) * Cdim + h * DH, Cdim);

    float m_s[8], l_s[8];
    v8f o0 = {}, o1 = {};
#pragma unroll
    for (int v = 0; v < 8; ++v) { m_s[v] = -1e30f; l_s[v] = 0.f; }
    const float scale = 0.17677669529663687f;   // 1/sqrt(32)

    // staging coords: 128 threads x 32B per tile
    int r = t >> 1, o = (t & 1) * 16;
    unsigned lk = (unsigned)((r * DH + o) * 2);
    unsigned lv = 4096u + lk;

    for (int nc = 0; nc < LK; nc += 64) {
        __syncthreads();   // previous iteration's tile reads done
        const h16* ks = KVp + (size_t)(nc + r) * 512 + h * DH + o;
        const h16* vs = ks + 256;
        async_ld128(lk,       ks);
        async_ld128(lk + 16u, ks + 8);
        async_ld128(lv,       vs);
        async_ld128(lv + 16u, vs + 8);
        if (nc + 64 < LK) __builtin_prefetch(ks + 64 * 512, 0, 1);
        async_wait0();
        __syncthreads();

        v8f s[4];
#pragma unroll
        for (int j = 0; j < 4; ++j) {
            v16h kb = load_bfrag_nt(Kt + j * 16 * DH, DH);
            v8f z = {};
            s[j] = wmma_f16(qf, kb, z);
        }
        float mx[8];
#pragma unroll
        for (int v = 0; v < 8; ++v) {
            s[0][v] *= scale; s[1][v] *= scale; s[2][v] *= scale; s[3][v] *= scale;
            mx[v] = fmaxf(fmaxf(s[0][v], s[1][v]), fmaxf(s[2][v], s[3][v]));
        }
        for (int d = 1; d < 16; d <<= 1)
#pragma unroll
            for (int v = 0; v < 8; ++v) mx[v] = fmaxf(mx[v], __shfl_xor(mx[v], d));
        float rs[8];
#pragma unroll
        for (int v = 0; v < 8; ++v) {
            float mn = fmaxf(m_s[v], mx[v]);
            float al = __expf(m_s[v] - mn);
            m_s[v] = mn;
            o0[v] *= al; o1[v] *= al; l_s[v] *= al;
            rs[v] = 0.f;
        }
#pragma unroll
        for (int j = 0; j < 4; ++j)
#pragma unroll
            for (int v = 0; v < 8; ++v) {
                float p = __expf(s[j][v] - m_s[v]);
                s[j][v] = p; rs[v] += p;
            }
        for (int d = 1; d < 16; d <<= 1)
#pragma unroll
            for (int v = 0; v < 8; ++v) rs[v] += __shfl_xor(rs[v], d);
#pragma unroll
        for (int v = 0; v < 8; ++v) l_s[v] += rs[v];

        // transpose P through per-wave LDS scratch into A-fragment layout
#pragma unroll
        for (int j = 0; j < 4; ++j)
#pragma unroll
            for (int v = 0; v < 8; ++v)
                ps[(v + 8 * hg) * 64 + j * 16 + lcol] = (h16)s[j][v];
        asm volatile("s_wait_dscnt 0x0" ::: "memory");
        v16h p0 = load_afrag(ps, 64);
        v16h p1 = load_afrag(ps + 32, 64);

        v16h vb;
        vb = load_bfrag_nn(Vt, DH);                o0 = wmma_f16(p0, vb, o0);
        vb = load_bfrag_nn(Vt + 16, DH);           o1 = wmma_f16(p0, vb, o1);
        vb = load_bfrag_nn(Vt + 32 * DH, DH);      o0 = wmma_f16(p1, vb, o0);
        vb = load_bfrag_nn(Vt + 32 * DH + 16, DH); o1 = wmma_f16(p1, vb, o1);
    }
#pragma unroll
    for (int v = 0; v < 8; ++v) {
        float inv = 1.f / l_s[v];
        int gm = qb + wv * 16 + v + 8 * hg;
        Oh[(size_t)gm * Cdim + h * DH + lcol]      = (h16)(o0[v] * inv);
        Oh[(size_t)gm * Cdim + h * DH + 16 + lcol] = (h16)(o1[v] * inv);
    }
}

// ---------------- host orchestration ----------------
extern "C" void kernel_launch(void* const* d_in, const int* in_sizes, int n_in,
                              void* d_out, int out_size, void* d_ws, size_t ws_size,
                              hipStream_t stream) {
    const float* x_ego  = (const float*)d_in[0];
    const float* x_agent= (const float*)d_in[1];
    const float* Wf1 = (const float*)d_in[2];  const float* bf1 = (const float*)d_in[3];
    const float* Wf2 = (const float*)d_in[4];  const float* bf2 = (const float*)d_in[5];
    const float* Wf3 = (const float*)d_in[6];  const float* bf3 = (const float*)d_in[7];
    const float* Wqkv= (const float*)d_in[8];  const float* bqkv= (const float*)d_in[9];
    const float* Wo  = (const float*)d_in[10]; const float* bo  = (const float*)d_in[11];
    const float* W1  = (const float*)d_in[12]; const float* b1  = (const float*)d_in[13];
    const float* W2  = (const float*)d_in[14]; const float* b2  = (const float*)d_in[15];
    const float* g1  = (const float*)d_in[16]; const float* be1 = (const float*)d_in[17];
    const float* g2  = (const float*)d_in[18]; const float* be2 = (const float*)d_in[19];
    const int* pos_ego   = (const int*)d_in[20];
    const int* pos_agent = (const int*)d_in[21];
    float* out = (float*)d_out;

    // ---- carve workspace ----
    char* base = (char*)d_ws;
    size_t off = 0;
    auto carve = [&](size_t bytes) -> void* {
        void* p = base + off;
        off = (off + bytes + 255) & ~(size_t)255;
        return p;
    };
    h16*  wWf1  = (h16*) carve((size_t)Cdim * 2 * Cdim * 2);
    h16*  wWf2  = (h16*) carve((size_t)Cdim * Cdim * 2);
    h16*  wWf3  = (h16*) carve((size_t)Cdim * Cdim * 2);
    h16*  wWqkv = (h16*) carve((size_t)3 * Cdim * Cdim * 2);
    h16*  wWo   = (h16*) carve((size_t)Cdim * Cdim * 2);
    h16*  wW1   = (h16*) carve((size_t)DF * Cdim * 2);
    h16*  wW2   = (h16*) carve((size_t)Cdim * DF * 2);
    h16*  fusedin = (h16*) carve((size_t)LE * 2 * Cdim * 2);
    h16*  h1    = (h16*) carve((size_t)LE * Cdim * 2);
    h16*  h2    = (h16*) carve((size_t)LE * Cdim * 2);
    float* fused3 = (float*) carve((size_t)LE * Cdim * 4);
    h16*  kvh   = (h16*) carve((size_t)LK * Cdim * 2);
    float* qf32 = (float*) carve((size_t)LQ * Cdim * 4);
    h16*  qh16  = (h16*) carve((size_t)LQ * Cdim * 2);
    h16*  Qp    = (h16*) carve((size_t)LQ * Cdim * 2);
    h16*  KVp   = (h16*) carve((size_t)LK * 2 * Cdim * 2);
    h16*  oh16  = (h16*) carve((size_t)LQ * Cdim * 2);
    float* oproj= (float*) carve((size_t)LQ * Cdim * 4);
    float* yqf  = (float*) carve((size_t)LQ * Cdim * 4);
    h16*  yqh   = (h16*) carve((size_t)LQ * Cdim * 2);
    h16*  ffh   = (h16*) carve((size_t)LQ * DF * 2);
    float* ff   = (float*) carve((size_t)LQ * Cdim * 4);
    int*  a4e   = (int*) carve(LE * 4);
    int*  ma    = (int*) carve(LA * 4);
    int*  ridx  = (int*) carve(LA * 4);
    (void)ws_size; (void)n_in; (void)in_sizes; (void)out_size;

    // ---- weight conversions to f16 ----
    auto cvt = [&](const float* s, h16* d, int n) {
        k_cvt<<<dim3((n + 255) / 256), dim3(256), 0, stream>>>(s, d, n);
    };
    cvt(Wf1, wWf1, Cdim * 2 * Cdim);
    cvt(Wf2, wWf2, Cdim * Cdim);
    cvt(Wf3, wWf3, Cdim * Cdim);
    cvt(Wqkv, wWqkv, 3 * Cdim * Cdim);
    cvt(Wo, wWo, Cdim * Cdim);
    cvt(W1, wW1, DF * Cdim);
    cvt(W2, wW2, Cdim * DF);

    // ---- matching + input staging ----
    k_match<<<dim3(LE), dim3(256), 0, stream>>>(x_ego, x_agent, pos_ego, pos_agent, a4e, fusedin);
    k_matched_a<<<dim3(LA / 256), dim3(256), 0, stream>>>(pos_ego, pos_agent, ma);
    k_remain<<<dim3(LA / 256), dim3(256), 0, stream>>>(ma, ridx);
    k_build_kv<<<dim3(LK * Cdim / 256), dim3(256), 0, stream>>>(x_ego, x_agent, kvh);

    // ---- fusion MLP (relu, relu, none) ----
    k_gemm_nt<<<dim3(LE / 128, Cdim / 64), dim3(256), GEMM_LDS_BYTES, stream>>>(
        fusedin, 2 * Cdim, wWf1, 2 * Cdim, bf1, nullptr, h1, Cdim, 2 * Cdim, 1);
    k_gemm_nt<<<dim3(LE / 128, Cdim / 64), dim3(256), GEMM_LDS_BYTES, stream>>>(
        h1, Cdim, wWf2, Cdim, bf2, nullptr, h2, Cdim, Cdim, 1);
    k_gemm_nt<<<dim3(LE / 128, Cdim / 64), dim3(256), GEMM_LDS_BYTES, stream>>>(
        h2, Cdim, wWf3, Cdim, bf3, fused3, nullptr, Cdim, Cdim, 0);

    // ---- build q (select fused vs ego, append remaining agents) ----
    k_build_q<<<dim3(LQ * Cdim / 256), dim3(256), 0, stream>>>(
        x_ego, x_agent, fused3, a4e, ridx, qf32, qh16);

    // ---- Q / KV projections ----
    k_gemm_nt<<<dim3(LQ / 128, Cdim / 64), dim3(256), GEMM_LDS_BYTES, stream>>>(
        qh16, Cdim, wWqkv, Cdim, bqkv, nullptr, Qp, Cdim, Cdim, 0);
    k_gemm_nt<<<dim3(LK / 128, (2 * Cdim) / 64), dim3(256), GEMM_LDS_BYTES, stream>>>(
        kvh, Cdim, wWqkv + (size_t)Cdim * Cdim, Cdim, bqkv + Cdim,
        nullptr, KVp, 2 * Cdim, Cdim, 0);

    // ---- attention (flash, online softmax) ----
    k_attn<<<dim3(LQ / 64, Hh), dim3(128), ATTN_LDS_BYTES, stream>>>(Qp, KVp, oh16);

    // ---- out-proj + residual LN1 ----
    k_gemm_nt<<<dim3(LQ / 128, Cdim / 64), dim3(256), GEMM_LDS_BYTES, stream>>>(
        oh16, Cdim, wWo, Cdim, bo, oproj, nullptr, Cdim, Cdim, 0);
    k_ln<<<dim3(LQ), dim3(Cdim), 0, stream>>>(qf32, oproj, g1, be1, yqf, yqh);

    // ---- FFN + residual LN2 -> out ----
    k_gemm_nt<<<dim3(LQ / 128, DF / 64), dim3(256), GEMM_LDS_BYTES, stream>>>(
        yqh, Cdim, wW1, Cdim, b1, nullptr, ffh, DF, Cdim, 1);
    k_gemm_nt<<<dim3(LQ / 128, Cdim / 64), dim3(256), GEMM_LDS_BYTES, stream>>>(
        ffh, DF, wW2, DF, b2, ff, nullptr, Cdim, DF, 0);
    k_ln<<<dim3(LQ), dim3(Cdim), 0, stream>>>(yqf, ff, g2, be2, out, nullptr);
}
